// FeatureWithRelativePosition_3178275799379
// MI455X (gfx1250) — compile-verified, matrix operand order checked by reading, not run on hardware
//
#include <hip/hip_runtime.h>
#include <hip/hip_bf16.h>

typedef _Float16 v16h __attribute__((ext_vector_type(16)));
typedef _Float16 v8h  __attribute__((ext_vector_type(8)));
typedef float    v8f  __attribute__((ext_vector_type(8)));

#define N_PTS 4096
#define FEAT  64
#define BATCH 4
#define TOTROWS (BATCH * N_PTS)      // 16384
#define KSPLIT 4
#define KTILE (N_PTS / KSPLIT)       // 1024
#define ROWS_PER_BLOCK 128           // 8 waves x 16 rows
#define LN_EPS 1e-5f

// ---- prologue: W fp32 -> f16 into workspace -------------------------------
__global__ void w_to_f16_kernel(const float* __restrict__ W,
                                _Float16* __restrict__ Wh, int n) {
  int i = blockIdx.x * blockDim.x + threadIdx.x;
  if (i < n) Wh[i] = (_Float16)W[i];
}

// ---- fused dist tile -> WMMA GEMM (optionally split-K) --------------------
// SPLIT=false: full K + inline LayerNorm/SiLU -> out
// SPLIT=true : K slice (blockIdx.y) -> fp32 partials in ws
template <bool WS_F16, bool SPLIT>
__global__ void __launch_bounds__(256) gemm_kernel(
    const float* __restrict__ pos,    // (BS, N, 3)
    const float* __restrict__ W,      // (FEAT, N) fp32
    const _Float16* __restrict__ Wh,  // (FEAT, N) f16 (valid if WS_F16)
    const float* __restrict__ bias,   // (FEAT)
    const float* __restrict__ gamma,  // (FEAT)
    const float* __restrict__ beta,   // (FEAT)
    float* __restrict__ dst)          // SPLIT ? partials : (BS,N,FEAT) out
{
  constexpr int KLEN = SPLIT ? KTILE : N_PTS;
  __shared__ float Xs[KLEN];          // SoA: runs of 8 floats -> 2x ds_load_b128
  __shared__ float Ys[KLEN];
  __shared__ float Zs[KLEN];

  const int b    = blockIdx.x / (N_PTS / ROWS_PER_BLOCK);
  const int rg   = blockIdx.x % (N_PTS / ROWS_PER_BLOCK);
  const int ks   = SPLIT ? blockIdx.y : 0;
  const int kBase = ks * KLEN;
  const int tid  = threadIdx.x;
  const int lane = tid & 31;
  const int wave = tid >> 5;
  const int l15  = lane & 15;
  const int g    = lane >> 4;         // half-wave group
  const int kg8  = 8 * g;

  // Stage this block's K-slice of points into LDS, SoA layout.
  const float* pb = pos + ((size_t)b * N_PTS + kBase) * 3;
  for (int m = tid; m < KLEN; m += 256) {
    Xs[m] = pb[m * 3 + 0];
    Ys[m] = pb[m * 3 + 1];
    Zs[m] = pb[m * 3 + 2];
  }
  __syncthreads();

  const int row0 = rg * ROWS_PER_BLOCK + wave * 16;   // within batch
  // This lane's A-row point (A layout: M = lane & 15 for both halves).
  const float* pr = pos + ((size_t)b * N_PTS + row0 + l15) * 3;
  const float px = pr[0], py = pr[1], pz = pr[2];
  v8f px8, py8, pz8;
#pragma unroll
  for (int e = 0; e < 8; ++e) { px8[e] = px; py8[e] = py; pz8[e] = pz; }

  // Accumulators: 4 feature groups of 16; bias folded into K-slice 0 only.
  v8f acc[4];
  float gam[4], bet[4];
#pragma unroll
  for (int fg = 0; fg < 4; ++fg) {
    const int f = fg * 16 + l15;
    const float bf = (SPLIT && ks != 0) ? 0.0f : bias[f];
#pragma unroll
    for (int v = 0; v < 8; ++v) acc[fg][v] = bf;
    if (!SPLIT) { gam[fg] = gamma[f]; bet[fg] = beta[f]; }
  }

  for (int k0 = 0; k0 < KLEN; k0 += 32) {
    // A fragment. Halves h=0..7 -> cols kg8+k0+0..7, h=8..15 -> +16..23:
    // two runs of 8 consecutive columns -> whole-run v8f LDS loads (b128 x2)
    // feeding packed v_pk_*_f32 math on adjacent register pairs.
    v16h A;
#pragma unroll
    for (int run = 0; run < 2; ++run) {
      const int c = k0 + kg8 + 16 * run;               // uniform per half
      const v8f qx = *(const v8f*)&Xs[c];
      const v8f qy = *(const v8f*)&Ys[c];
      const v8f qz = *(const v8f*)&Zs[c];
      const v8f dx = px8 - qx;                          // v_pk_add_f32 (neg)
      const v8f dy = py8 - qy;
      const v8f dz = pz8 - qz;
      const v8f s  = dx * dx + dy * dy + dz * dz;       // pk_mul + 2x pk_fma
#pragma unroll
      for (int e = 0; e < 8; ++e)
        A[8 * run + e] = (_Float16)__builtin_amdgcn_sqrtf(s[e]);
    }
#pragma unroll
    for (int fg = 0; fg < 4; ++fg) {
      const int f = fg * 16 + l15;                      // B layout: N = l15
      v16h B;
      if (WS_F16) {
        const _Float16* wp = Wh + (size_t)f * N_PTS + kBase + k0 + kg8;
        const v8h lo = *(const v8h*)(wp);               // K = 8g + 0..7
        const v8h hi = *(const v8h*)(wp + 16);          // K = 8g + 16..23
        B = __builtin_shufflevector(lo, hi, 0, 1, 2, 3, 4, 5, 6, 7,
                                    8, 9, 10, 11, 12, 13, 14, 15);
      } else {
        const float* wp = W + (size_t)f * N_PTS + kBase + k0 + kg8;
#pragma unroll
        for (int h = 0; h < 8; ++h) B[h] = (_Float16)wp[h];
#pragma unroll
        for (int h = 0; h < 8; ++h) B[8 + h] = (_Float16)wp[16 + h];
      }
      acc[fg] = __builtin_amdgcn_wmma_f32_16x16x32_f16(
          false, A, false, B, (short)0, acc[fg], false, false);
    }
  }

  if (SPLIT) {
    // Store fp32 partials: part[ks][b*N + row][feat]
    const size_t base =
        ((size_t)ks * TOTROWS + (size_t)b * N_PTS + row0) * FEAT;
#pragma unroll
    for (int v = 0; v < 8; ++v) {
      const int row = v + 8 * g;
#pragma unroll
      for (int fg = 0; fg < 4; ++fg)
        dst[base + (size_t)row * FEAT + fg * 16 + l15] = acc[fg][v];
    }
  } else {
    // Inline LayerNorm(64) + SiLU. Rows = v + 8g, feats striped on l15.
    const size_t outBase = ((size_t)b * N_PTS + row0) * FEAT;
#pragma unroll
    for (int v = 0; v < 8; ++v) {
      float a0 = acc[0][v], a1 = acc[1][v], a2 = acc[2][v], a3 = acc[3][v];
      float s = a0 + a1 + a2 + a3;
      float q = a0 * a0 + a1 * a1 + a2 * a2 + a3 * a3;
#pragma unroll
      for (int off = 1; off < 16; off <<= 1) {          // stays within half
        s += __shfl_xor(s, off, 32);
        q += __shfl_xor(q, off, 32);
      }
      const float mu   = s * (1.0f / 64.0f);
      const float var  = q * (1.0f / 64.0f) - mu * mu;
      const float rinv = __builtin_amdgcn_rsqf(var + LN_EPS);
      const int   row  = v + 8 * g;
#pragma unroll
      for (int fg = 0; fg < 4; ++fg) {
        const float xn = (acc[fg][v] - mu) * rinv * gam[fg] + bet[fg];
        const float y  = xn * __builtin_amdgcn_rcpf(1.0f + __expf(-xn));
        dst[outBase + (size_t)row * FEAT + fg * 16 + l15] = y;
      }
    }
  }
}

// ---- finalize: reduce KSPLIT partials + LayerNorm + SiLU ------------------
__global__ void __launch_bounds__(256) finalize_kernel(
    const float* __restrict__ part,   // (KSPLIT, TOTROWS, FEAT)
    const float* __restrict__ gamma,
    const float* __restrict__ beta,
    float* __restrict__ out)          // (TOTROWS, FEAT)
{
  const int  wave = threadIdx.x >> 5;
  const int  lane = threadIdx.x & 31;
  const size_t row = (size_t)blockIdx.x * 8 + wave;    // one wave per row
  float x0 = 0.0f, x1 = 0.0f;                          // feats lane, lane+32
#pragma unroll
  for (int ks = 0; ks < KSPLIT; ++ks) {
    const float* p = part + ((size_t)ks * TOTROWS + row) * FEAT;
    x0 += p[lane];
    x1 += p[lane + 32];
  }
  float s = x0 + x1;
  float q = x0 * x0 + x1 * x1;
#pragma unroll
  for (int off = 1; off < 32; off <<= 1) {             // full-wave butterfly
    s += __shfl_xor(s, off, 32);
    q += __shfl_xor(q, off, 32);
  }
  const float mu   = s * (1.0f / 64.0f);
  const float var  = q * (1.0f / 64.0f) - mu * mu;
  const float rinv = __builtin_amdgcn_rsqf(var + LN_EPS);
  const float xn0  = (x0 - mu) * rinv * gamma[lane] + beta[lane];
  const float xn1  = (x1 - mu) * rinv * gamma[lane + 32] + beta[lane + 32];
  out[row * FEAT + lane]      = xn0 * __builtin_amdgcn_rcpf(1.0f + __expf(-xn0));
  out[row * FEAT + lane + 32] = xn1 * __builtin_amdgcn_rcpf(1.0f + __expf(-xn1));
}

extern "C" void kernel_launch(void* const* d_in, const int* in_sizes, int n_in,
                              void* d_out, int out_size, void* d_ws, size_t ws_size,
                              hipStream_t stream) {
  const float* pos   = (const float*)d_in[0]; // (4,4096,3)
  const float* W     = (const float*)d_in[1]; // (64,4096)
  const float* bias  = (const float*)d_in[2]; // (64,)
  const float* gamma = (const float*)d_in[3]; // (64,)
  const float* beta  = (const float*)d_in[4]; // (64,)
  float* out = (float*)d_out;

  const size_t whBytes   = (size_t)FEAT * N_PTS * sizeof(_Float16);         // 512 KB
  const size_t partBytes = (size_t)KSPLIT * TOTROWS * FEAT * sizeof(float); // 16 MB
  const int    nW        = FEAT * N_PTS;
  const dim3   blk(256);

  if (ws_size >= whBytes + partBytes) {
    // Best path: f16 W + split-K (4096 waves) + finalize pass.
    _Float16* Wh   = (_Float16*)d_ws;
    float*    part = (float*)((char*)d_ws + whBytes);
    w_to_f16_kernel<<<(nW + 255) / 256, 256, 0, stream>>>(W, Wh, nW);
    gemm_kernel<true, true>
        <<<dim3(BATCH * (N_PTS / ROWS_PER_BLOCK), KSPLIT), blk, 0, stream>>>(
            pos, W, Wh, bias, gamma, beta, part);
    finalize_kernel<<<TOTROWS / 8, blk, 0, stream>>>(part, gamma, beta, out);
  } else if (ws_size >= whBytes) {
    _Float16* Wh = (_Float16*)d_ws;
    w_to_f16_kernel<<<(nW + 255) / 256, 256, 0, stream>>>(W, Wh, nW);
    gemm_kernel<true, false>
        <<<dim3(BATCH * (N_PTS / ROWS_PER_BLOCK)), blk, 0, stream>>>(
            pos, W, Wh, bias, gamma, beta, out);
  } else {
    gemm_kernel<false, false>
        <<<dim3(BATCH * (N_PTS / ROWS_PER_BLOCK)), blk, 0, stream>>>(
            pos, W, nullptr, bias, gamma, beta, out);
  }
}
